// MultiHopGATv2_51032801411299
// MI455X (gfx1250) — matrix operand order repaired and checked by reading, not compiled, not run on hardware
//
#include <hip/hip_runtime.h>
#include <hip/hip_bf16.h>

// ---------------------------------------------------------------------------
// Types for CDNA5 WMMA (wave32): 16x16x32 bf16 -> f32
// ---------------------------------------------------------------------------
typedef __bf16 v16bf __attribute__((ext_vector_type(16)));
typedef __bf16 v8bf  __attribute__((ext_vector_type(8)));
typedef float  v8f   __attribute__((ext_vector_type(8)));

#define HEADS 4
#define NEG_SLOPE 0.2f
#define NT 2   // N-tiles per wave in the GEMM

// ---------------------------------------------------------------------------
// Ordered-float <-> uint monotone encoding for atomicMax on floats
// ---------------------------------------------------------------------------
__device__ __forceinline__ unsigned f2o(float f) {
    unsigned u = __float_as_uint(f);
    return (u & 0x80000000u) ? ~u : (u | 0x80000000u);
}
__device__ __forceinline__ float o2f(unsigned v) {
    unsigned u = (v & 0x80000000u) ? (v & 0x7fffffffu) : ~v;
    return __uint_as_float(u);
}

// ---------------------------------------------------------------------------
// Fill kernel (d_ws is poisoned by the harness; clear accumulators each call)
// ---------------------------------------------------------------------------
__global__ void __launch_bounds__(256) fill_u32(unsigned* __restrict__ p,
                                                unsigned v, long long n) {
    long long i = (long long)blockIdx.x * blockDim.x + threadIdx.x;
    if (i < n) p[i] = v;
}

// ---------------------------------------------------------------------------
// Pack f32 activations [N, Fin] -> bf16 [Npad, Kpad], zero padded.
// ---------------------------------------------------------------------------
__global__ void __launch_bounds__(256) pack_act_bf16(
    const float* __restrict__ in, __bf16* __restrict__ outb,
    int N, int Fin, int Npad, int Kpad)
{
    long long t = (long long)blockIdx.x * blockDim.x + threadIdx.x;
    long long total = (long long)Npad * Kpad;
    if (t >= total) return;
    int n = (int)(t / Kpad);
    int k = (int)(t % Kpad);
    float v = (n < N && k < Fin) ? in[(size_t)n * Fin + k] : 0.0f;
    outb[t] = (__bf16)v;
}

// ---------------------------------------------------------------------------
// Pack + transpose weights f32 [Fin, Wout] -> bf16 [Wout, Kpad], zero padded.
// ---------------------------------------------------------------------------
__global__ void __launch_bounds__(256) pack_wT_bf16(
    const float* __restrict__ W, __bf16* __restrict__ outb,
    int Fin, int Wout, int Kpad)
{
    long long t = (long long)blockIdx.x * blockDim.x + threadIdx.x;
    long long total = (long long)Wout * Kpad;
    if (t >= total) return;
    int n = (int)(t / Kpad);
    int k = (int)(t % Kpad);
    float v = (k < Fin) ? W[(size_t)k * Wout + n] : 0.0f;
    outb[t] = (__bf16)v;
}

// ---------------------------------------------------------------------------
// WMMA GEMM: out[m,n] = sum_k A[m,k] * Wt[n,k] + bias[n]
// A:  bf16 [Npad, Kpad] (zero padded)      -> 2x contiguous 16B loads/lane
// Wt: bf16 [Wout, Kpad] (transposed, pad)  -> 2x contiguous 16B loads/lane
// One 16x16 M-tile x NT N-tiles per wave, 8 waves/block.
// Wout must be a multiple of 16*NT*8 * gridDim.y selection (we use Wout%256==0).
// ---------------------------------------------------------------------------
__global__ void __launch_bounds__(256) gemm_wmma_bf16(
    const __bf16* __restrict__ Abf, const __bf16* __restrict__ Bt,
    const float* __restrict__ bias, float* __restrict__ out,
    int N, int Kpad, int Wout)
{
    const int lane   = threadIdx.x & 31;
    const int wave   = threadIdx.x >> 5;
    const int mtile  = blockIdx.x;
    const int ntile0 = (blockIdx.y * 8 + wave) * NT;
    const int l15    = lane & 15;
    const int hi     = lane >> 4;

    const int mA  = mtile * 16 + l15;   // A row (padded buffer, always valid)
    const int kbA = hi * 8;             // A K sub-base (ISA 16-bit A layout)
    const int kbB = hi * 8;             // B K sub-base (same K order as A)

    v8f acc[NT];
    #pragma unroll
    for (int j = 0; j < NT; ++j) acc[j] = (v8f){};

    const int ksteps = Kpad >> 5;
    for (int kt = 0; kt < ksteps; ++kt) {
        const int k0 = kt * 32;
        // A fragment: K = {k0+kbA..+7} and {k0+16+kbA..+7}, contiguous bf16.
        const __bf16* ap = Abf + (size_t)mA * Kpad + k0 + kbA;
        v8bf alo = *(const v8bf*)(ap);
        v8bf ahi = *(const v8bf*)(ap + 16);
        v16bf a;
        #pragma unroll
        for (int i = 0; i < 8; ++i) { a[i] = alo[i]; a[8 + i] = ahi[i]; }

        #pragma unroll
        for (int j = 0; j < NT; ++j) {
            const int n = (ntile0 + j) * 16 + l15;
            // B fragment: K = {k0+kbB..+7} and {k0+16+kbB..+7}, contiguous bf16.
            const __bf16* bp = Bt + (size_t)n * Kpad + k0 + kbB;
            v8bf b0 = *(const v8bf*)(bp);
            v8bf b1 = *(const v8bf*)(bp + 16);
            v16bf b;
            #pragma unroll
            for (int i = 0; i < 8; ++i) { b[i] = b0[i]; b[8 + i] = b1[i]; }
            acc[j] = __builtin_amdgcn_wmma_f32_16x16x32_bf16(
                false, a, false, b, (short)0, acc[j], false, false);
        }
    }

    const int mbase = mtile * 16 + (hi << 3);   // D VGPR r -> row mbase + r
    #pragma unroll
    for (int j = 0; j < NT; ++j) {
        const int n = (ntile0 + j) * 16 + l15;
        const float bn = bias[n];
        #pragma unroll
        for (int r = 0; r < 8; ++r) {
            int m = mbase + r;
            if (m < N) out[(size_t)m * Wout + n] = acc[j][r] + bn;
        }
    }
}

// ---------------------------------------------------------------------------
// Edge logits: one thread per (edge, head).
// logit = sum_c att[h,c] * leaky_relu(xl[src,h,c] + xr[dst,h,c])
// Segment-max into maxb[dst,h] via ordered-uint atomicMax.
// Edges e >= E are self-loops (src = dst = e - E).
// ---------------------------------------------------------------------------
__global__ void __launch_bounds__(256) edge_logits(
    const float* __restrict__ xl, const float* __restrict__ xr,
    const float* __restrict__ att, const int* __restrict__ src,
    const int* __restrict__ dst, int E, int N, int C,
    float* __restrict__ logits, unsigned* __restrict__ maxb)
{
    long long t = (long long)blockIdx.x * blockDim.x + threadIdx.x;
    long long total = (long long)(E + N) * HEADS;
    if (t >= total) return;
    int e = (int)(t >> 2);
    int h = (int)(t & 3);
    int s, d;
    if (e < E) { s = src[e]; d = dst[e]; } else { s = d = e - E; }

    const int HC = HEADS * C;
    const float* pl = xl + (size_t)s * HC + h * C;
    const float* pr = xr + (size_t)d * HC + h * C;
    const float* pa = att + h * C;

    float acc = 0.0f;
    for (int c = 0; c < C; c += 4) {
        float4 vl = *(const float4*)(pl + c);
        float4 vr = *(const float4*)(pr + c);
        float4 va = *(const float4*)(pa + c);
        float v;
        v = vl.x + vr.x; v = v > 0.0f ? v : NEG_SLOPE * v; acc += va.x * v;
        v = vl.y + vr.y; v = v > 0.0f ? v : NEG_SLOPE * v; acc += va.y * v;
        v = vl.z + vr.z; v = v > 0.0f ? v : NEG_SLOPE * v; acc += va.z * v;
        v = vl.w + vr.w; v = v > 0.0f ? v : NEG_SLOPE * v; acc += va.w * v;
    }
    logits[t] = acc;
    atomicMax(&maxb[(size_t)d * HEADS + h], f2o(acc));
}

// ---------------------------------------------------------------------------
// Edge exp: a = exp(logit - max[dst,h]) (in place), denom[dst,h] += a
// ---------------------------------------------------------------------------
__global__ void __launch_bounds__(256) edge_exp(
    float* __restrict__ logits, const unsigned* __restrict__ maxb,
    float* __restrict__ denom, const int* __restrict__ dst, int E, int N)
{
    long long t = (long long)blockIdx.x * blockDim.x + threadIdx.x;
    long long total = (long long)(E + N) * HEADS;
    if (t >= total) return;
    int e = (int)(t >> 2);
    int h = (int)(t & 3);
    int d = (e < E) ? dst[e] : (e - E);
    float m = o2f(maxb[(size_t)d * HEADS + h]);
    float a = __expf(logits[t] - m);
    logits[t] = a;
    atomicAdd(&denom[(size_t)d * HEADS + h], a);
}

// ---------------------------------------------------------------------------
// Edge aggregate: agg[dst,h,c] += (a[e,h]/denom[dst,h]) * xl[src,h,c]
// One thread per (edge, 4-channel group): float4 gather + 4 f32 atomics.
// ---------------------------------------------------------------------------
__global__ void __launch_bounds__(256) edge_aggregate(
    const float* __restrict__ xl, const float* __restrict__ a,
    const float* __restrict__ denom, const int* __restrict__ src,
    const int* __restrict__ dst, int E, int N, int C,
    float* __restrict__ agg)
{
    long long t = (long long)blockIdx.x * blockDim.x + threadIdx.x;
    long long total = (long long)(E + N) * C;   // (HC/4) groups per edge == C
    if (t >= total) return;
    int e = (int)(t / C);
    int q = (int)(t % C);
    int j = q * 4;                // channel offset within [0, HC)
    int h = j / C;                // head index
    int s, d;
    if (e < E) { s = src[e]; d = dst[e]; } else { s = d = e - E; }

    const int HC = HEADS * C;
    float alpha = a[(size_t)e * HEADS + h] / denom[(size_t)d * HEADS + h];
    float4 v = *(const float4*)(xl + (size_t)s * HC + j);
    float* o = agg + (size_t)d * HC + j;
    atomicAdd(o + 0, alpha * v.x);
    atomicAdd(o + 1, alpha * v.y);
    atomicAdd(o + 2, alpha * v.z);
    atomicAdd(o + 3, alpha * v.w);
}

// ---------------------------------------------------------------------------
// Finalize: out[n,c] = mean_h agg[n,h,c] + bias[c], optional ReLU
// ---------------------------------------------------------------------------
__global__ void __launch_bounds__(256) finalize_mean_bias(
    const float* __restrict__ agg, const float* __restrict__ bias,
    float* __restrict__ out, int N, int C, int do_relu)
{
    long long t = (long long)blockIdx.x * blockDim.x + threadIdx.x;
    long long total = (long long)N * C;
    if (t >= total) return;
    int n = (int)(t / C);
    int c = (int)(t % C);
    const float* p = agg + (size_t)n * HEADS * C + c;
    float s = 0.25f * (p[0] + p[C] + p[2 * C] + p[3 * C]);
    s += bias[c];
    if (do_relu) s = s > 0.0f ? s : 0.0f;
    out[t] = s;
}

// ---------------------------------------------------------------------------
// Launch
// ---------------------------------------------------------------------------
static inline long long divup_ll(long long a, long long b) { return (a + b - 1) / b; }

extern "C" void kernel_launch(void* const* d_in, const int* in_sizes, int n_in,
                              void* d_out, int out_size, void* d_ws, size_t ws_size,
                              hipStream_t stream) {
    const float* x  = (const float*)d_in[0];
    const int*   ei = (const int*)d_in[1];
    const int N = in_sizes[0] / 9;
    const int E = in_sizes[1] / 2;
    const int Npad = (N + 15) & ~15;
    const int* src = ei;
    const int* dst = ei + E;

    const float *Wl[3], *bl[3], *Wr[3], *br[3], *attp[3], *bb[3];
    for (int i = 0; i < 3; ++i) {
        int base = 2 + i * 6;
        Wl[i]   = (const float*)d_in[base + 0];
        bl[i]   = (const float*)d_in[base + 1];
        Wr[i]   = (const float*)d_in[base + 2];
        br[i]   = (const float*)d_in[base + 3];
        attp[i] = (const float*)d_in[base + 4];
        bb[i]   = (const float*)d_in[base + 5];
    }

    // Workspace layout (256B aligned slabs)
    char* cur = (char*)d_ws;
    auto alloc = [&](size_t bytes) -> void* {
        void* p = (void*)cur;
        cur += (bytes + 255) & ~(size_t)255;
        return p;
    };
    const size_t HCmax = 512;
    float*    XL   = (float*)alloc((size_t)N * HCmax * 4);
    float*    XR   = (float*)alloc((size_t)N * HCmax * 4);
    float*    AGG  = (float*)alloc((size_t)N * HCmax * 4);
    float*    ACT  = (float*)alloc((size_t)N * 128 * 4);
    float*    LOG  = (float*)alloc((size_t)(E + N) * HEADS * 4);
    float*    DEN  = (float*)alloc((size_t)N * HEADS * 4);
    unsigned* MAXB = (unsigned*)alloc((size_t)N * HEADS * 4);
    __bf16*   ACTB = (__bf16*)alloc((size_t)Npad * 64 * 2);
    __bf16*   WLT  = (__bf16*)alloc((size_t)512 * 64 * 2);
    __bf16*   WRT  = (__bf16*)alloc((size_t)512 * 64 * 2);

    const int Cs[3] = {64, 64, 128};
    const float* act = x;
    int Fin = 9;

    const dim3 blk(256);
    const long long edgeH = (long long)(E + N) * HEADS;

    for (int l = 0; l < 3; ++l) {
        const int C    = Cs[l];
        const int HC   = HEADS * C;
        const int Kpad = (Fin <= 32) ? 32 : 64;

        // --- pack activations (bf16, zero padded) ---
        long long nPack = (long long)Npad * Kpad;
        pack_act_bf16<<<(unsigned)divup_ll(nPack, 256), blk, 0, stream>>>(
            act, ACTB, N, Fin, Npad, Kpad);

        // --- pack + transpose weights (bf16, zero padded) ---
        long long nW = (long long)HC * Kpad;
        pack_wT_bf16<<<(unsigned)divup_ll(nW, 256), blk, 0, stream>>>(
            Wl[l], WLT, Fin, HC, Kpad);
        pack_wT_bf16<<<(unsigned)divup_ll(nW, 256), blk, 0, stream>>>(
            Wr[l], WRT, Fin, HC, Kpad);

        // --- dense transforms (WMMA) ---
        dim3 gg((unsigned)(Npad / 16), (unsigned)(HC / (16 * NT * 8)));
        gemm_wmma_bf16<<<gg, blk, 0, stream>>>(ACTB, WLT, bl[l], XL, N, Kpad, HC);
        gemm_wmma_bf16<<<gg, blk, 0, stream>>>(ACTB, WRT, br[l], XR, N, Kpad, HC);

        // --- clear accumulators ---
        long long nMax = (long long)N * HEADS;
        long long nAgg = (long long)N * HC;
        fill_u32<<<(unsigned)divup_ll(nMax, 256), blk, 0, stream>>>(MAXB, 0u, nMax);
        fill_u32<<<(unsigned)divup_ll(nMax, 256), blk, 0, stream>>>((unsigned*)DEN, 0u, nMax);
        fill_u32<<<(unsigned)divup_ll(nAgg, 256), blk, 0, stream>>>((unsigned*)AGG, 0u, nAgg);

        // --- edge phase: logits + segment max ---
        edge_logits<<<(unsigned)divup_ll(edgeH, 256), blk, 0, stream>>>(
            XL, XR, attp[l], src, dst, E, N, C, LOG, MAXB);

        // --- exp + segment sum ---
        edge_exp<<<(unsigned)divup_ll(edgeH, 256), blk, 0, stream>>>(
            LOG, MAXB, DEN, dst, E, N);

        // --- weighted scatter-add ---
        long long nAggT = (long long)(E + N) * C;
        edge_aggregate<<<(unsigned)divup_ll(nAggT, 256), blk, 0, stream>>>(
            XL, LOG, DEN, src, dst, E, N, C, AGG);

        // --- head mean + bias (+ReLU) ---
        long long nFin = (long long)N * C;
        float* outp = (l == 2) ? (float*)d_out : ACT;
        finalize_mean_bias<<<(unsigned)divup_ll(nFin, 256), blk, 0, stream>>>(
            AGG, bb[l], outp, N, C, (l < 2) ? 1 : 0);

        act = ACT;
        Fin = C;
    }
}
